// SocialLSTM_55095840473278
// MI455X (gfx1250) — matrix operand order repaired and checked by reading, not compiled, not run on hardware
//
#include <hip/hip_runtime.h>
#include <hip/hip_bf16.h>

// ---------------- problem constants ----------------
#define TT 20
#define NN 512
#define G2C 16
#define RR 128
#define EE 64
#define II 2
#define OO 5

typedef __attribute__((ext_vector_type(16))) __bf16 v16bf;
typedef __attribute__((ext_vector_type(8)))  __bf16 v8bf;
typedef __attribute__((ext_vector_type(8)))  float  v8f;
typedef __attribute__((ext_vector_type(4)))  float  v4f;

__device__ __forceinline__ v16bf frag16(const __bf16* p0, const __bf16* p1) {
  union { v16bf v; v8bf h[2]; } u;
  u.h[0] = *(const v8bf*)p0;
  u.h[1] = *(const v8bf*)p1;
  return u.v;
}

__device__ __forceinline__ v8f wmma_bf16(v16bf a, v16bf b, v8f c) {
  // D = A(16x32 bf16) * B(32x16 bf16) + C(16x16 f32)
  return __builtin_amdgcn_wmma_f32_16x16x32_bf16(
      /*neg_a=*/false, a, /*neg_b=*/false, b,
      /*c_mod=*/(short)0, c, /*reuse_a=*/false, /*reuse_b=*/false);
}

__device__ __forceinline__ float sigm(float v) {
  return 1.0f / (1.0f + __expf(-v));
}

// ---------------------------------------------------------------------------
// Kernel 0: one-time packing of bf16 operands + state init.
// ---------------------------------------------------------------------------
__global__ void init_kernel(const float* __restrict__ hidden0,
                            const float* __restrict__ cell0,
                            const float* __restrict__ W_ih,   // [4R, 2E]
                            const float* __restrict__ W_hh,   // [4R, R]
                            const float* __restrict__ W_t,    // [E, G2*R]
                            float*  __restrict__ c_ws,        // [N, R]
                            __bf16* __restrict__ hrow,        // [N, R]
                            __bf16* __restrict__ hT,          // [R, N]
                            __bf16* __restrict__ Wcat,        // [4R, 2E + R]
                            __bf16* __restrict__ Wt) {        // [E, G2*R]
  const int idx = blockIdx.x * blockDim.x + threadIdx.x;
  const int stride = gridDim.x * blockDim.x;
  for (int i = idx; i < NN * RR; i += stride) {
    const float h = hidden0[i];
    c_ws[i] = cell0[i];
    hrow[i] = (__bf16)h;
    const int n = i / RR, r = i % RR;
    hT[r * NN + n] = (__bf16)h;
  }
  for (int i = idx; i < 4 * RR * 2 * EE; i += stride) {       // W_ih part
    const int j = i / (2 * EE), k = i % (2 * EE);
    Wcat[j * (2 * EE + RR) + k] = (__bf16)W_ih[i];
  }
  for (int i = idx; i < 4 * RR * RR; i += stride) {           // W_hh part
    const int j = i / RR, k = i % RR;
    Wcat[j * (2 * EE + RR) + 2 * EE + k] = (__bf16)W_hh[i];
  }
  for (int i = idx; i < EE * G2C * RR; i += stride)
    Wt[i] = (__bf16)W_t[i];
}

// ---------------------------------------------------------------------------
// Kernel 1: social[n, g*R + r] = sum_m grid[n,m,g] * h[m,r]
// 2 nodes per block (256 threads / 8 waves): waves 0-3 -> node n0,
// waves 4-7 -> node n1. Per node, wave owns r-tiles [32w,+16) and [32w+16,+16).
// A tile (16g x 32m) is transposed into LDS as bf16, DOUBLE-BUFFERED so
// staging chunk k+1 overlaps the WMMAs of chunk k (one barrier per chunk).
// B fragments come straight from the L2-resident hT [R,N].
// ---------------------------------------------------------------------------
__device__ __forceinline__ void stageA(__bf16 (*dst)[40],
                                       const float* __restrict__ gbase,
                                       int m0, int t128) {
  const int mloc = t128 >> 2;
  const int g0   = (t128 & 3) * 4;
  v4f g4 = *(const v4f*)(gbase + (size_t)(m0 + mloc) * G2C + g0);
  dst[g0 + 0][mloc] = (__bf16)g4.x;
  dst[g0 + 1][mloc] = (__bf16)g4.y;
  dst[g0 + 2][mloc] = (__bf16)g4.z;
  dst[g0 + 3][mloc] = (__bf16)g4.w;
}

__global__ void __launch_bounds__(256) social_wmma_kernel(
    const float*  __restrict__ grid_t,   // [N, N, G2] (this timestep)
    const __bf16* __restrict__ hT,       // [R, N]
    __bf16*       __restrict__ social) { // [N, G2*R]
  __shared__ __bf16 At[2][2][16][40];    // [buf][node-half][g][m], 80B rows
  const int tid  = threadIdx.x;
  const int half = tid >> 7;             // node within block
  const int n    = blockIdx.x * 2 + half;
  const int t128 = tid & 127;
  const int wave = t128 >> 5;            // 0..3 within node-half
  const int lane = tid & 31;
  const int l15  = lane & 15;
  const bool hiL = lane >= 16;
  const int r0   = wave * 32;

  const float* gbase = grid_t + (size_t)n * NN * G2C;

  v8f acc0 = {};
  v8f acc1 = {};

  stageA(At[0][half], gbase, 0, t128);   // prologue: stage chunk 0
  __syncthreads();

  for (int kc = 0; kc < NN / 32; ++kc) {
    const int m0 = kc * 32;
    const __bf16 (*Ab)[40] = At[kc & 1][half];
    // A: lane M=l15, 16 consecutive K (lanes 0-15: K 0..15; 16-31: K 16..31)
    v16bf a = frag16(&Ab[l15][hiL ? 16 : 0], &Ab[l15][hiL ? 24 : 8]);
    if (kc + 1 < NN / 32)                // overlap: stage next chunk
      stageA(At[(kc + 1) & 1][half], gbase, m0 + 32, t128);
    if (kc + 4 < NN / 32)                // stream-prefetch 8KB ahead
      __builtin_prefetch(gbase + (size_t)(m0 + 128 + (t128 >> 2)) * G2C, 0, 1);
    // B: lane N=l15, 16 consecutive K (lanes 0-15: K 0..15; 16-31: K 16..31)
    const __bf16* hr0 = hT + (size_t)(r0 + l15) * NN + m0 + (hiL ? 16 : 0);
    const __bf16* hr1 = hT + (size_t)(r0 + 16 + l15) * NN + m0 + (hiL ? 16 : 0);
    v16bf b0 = frag16(hr0, hr0 + 8);
    v16bf b1 = frag16(hr1, hr1 + 8);
    acc0 = wmma_bf16(a, b0, acc0);
    acc1 = wmma_bf16(a, b1, acc1);
    __syncthreads();                     // next buffer staged; this one free
  }

  __bf16* srow = social + (size_t)n * (G2C * RR);
#pragma unroll
  for (int i = 0; i < 8; ++i) {
    const int g = i + (hiL ? 8 : 0);
    srow[g * RR + r0 + l15]      = (__bf16)acc0[i];
    srow[g * RR + r0 + 16 + l15] = (__bf16)acc1[i];
  }
}

// ---------------------------------------------------------------------------
// Kernel 2: te = relu(social @ W_tensor^T + b_tensor)  [512 x 2048 x 64 GEMM]
//           ie = relu(nodes_t @ W_in^T + b_in)         (K=2, plain VALU)
// x[n, 0:64] = ie, x[n, 64:128] = te. One block per 16-node tile, 4 waves,
// wave w owns e-tile [16w, 16w+16).
// ---------------------------------------------------------------------------
__global__ void __launch_bounds__(128) embed_kernel(
    const __bf16* __restrict__ social,   // [N, 2048]
    const __bf16* __restrict__ Wt,       // [E, 2048]
    const float*  __restrict__ b_tensor, // [E]
    const float*  __restrict__ nodes_t,  // [N, I]
    const float*  __restrict__ W_in,     // [E, I]
    const float*  __restrict__ b_in,     // [E]
    __bf16*       __restrict__ x) {      // [N, 2E]
  const int nBase = blockIdx.x * 16;
  const int tid   = threadIdx.x;
  const int wave  = tid >> 5;
  const int lane  = tid & 31;
  const int l15   = lane & 15;
  const bool hiL  = lane >= 16;
  const int e0    = wave * 16;

  v8f acc = {};
  const __bf16* arow = social + (size_t)(nBase + l15) * (G2C * RR);
  const __bf16* brow = Wt + (size_t)(e0 + l15) * (G2C * RR);
  for (int kc = 0; kc < (G2C * RR) / 32; ++kc) {
    const int k = kc * 32;
    v16bf a = frag16(arow + k + (hiL ? 16 : 0), arow + k + (hiL ? 24 : 8));
    const __bf16* bp = brow + k + (hiL ? 16 : 0);
    v16bf b = frag16(bp, bp + 8);
    acc = wmma_bf16(a, b, acc);
  }
  const int e = e0 + l15;
  const float bias = b_tensor[e];
#pragma unroll
  for (int i = 0; i < 8; ++i) {
    const int nl = i + (hiL ? 8 : 0);
    float v = acc[i] + bias;
    v = v > 0.f ? v : 0.f;
    x[(size_t)(nBase + nl) * (2 * EE) + EE + e] = (__bf16)v;
  }
  {   // input embedding, 8 outputs per thread
    const int nl = tid >> 3;
    const int eb = (tid & 7) * 8;
    const int n = nBase + nl;
    const float x0 = nodes_t[n * II + 0];
    const float x1 = nodes_t[n * II + 1];
#pragma unroll
    for (int j = 0; j < 8; ++j) {
      const int e2 = eb + j;
      float v = x0 * W_in[e2 * II + 0] + x1 * W_in[e2 * II + 1] + b_in[e2];
      v = v > 0.f ? v : 0.f;
      x[(size_t)n * (2 * EE) + e2] = (__bf16)v;
    }
  }
}

// ---------------------------------------------------------------------------
// Kernel 3: gates = [x|h] @ [W_ih|W_hh]^T + b  ->  LSTM pointwise  ->  out.
// One block per 16-node tile; wave w computes gate columns [128w, 128w+128)
// (exactly gate w in torch i,f,g,o order). K = 256 over concatenated x|h.
// Each block reads/writes only its own 16 rows of h/c (no cross-block hazard).
// ---------------------------------------------------------------------------
__global__ void __launch_bounds__(128) lstm_kernel(
    const __bf16* __restrict__ x,        // [N, 2E]
    const __bf16* __restrict__ hrow,     // [N, R] prev h (bf16)
    const __bf16* __restrict__ Wcat,     // [4R, 2E+R]
    const float*  __restrict__ b_ih,
    const float*  __restrict__ b_hh,
    float*        __restrict__ c,        // [N, R] f32 state
    const float*  __restrict__ W_out,    // [O, R]
    const float*  __restrict__ b_out,    // [O]
    float*        __restrict__ out_t,    // [N, O] slice of d_out
    __bf16*       __restrict__ hrow_new, // [N, R]
    __bf16*       __restrict__ hT_new,   // [R, N]
    float*        __restrict__ h_final,  // nullptr except last step
    float*        __restrict__ c_final) {
  __shared__ float G[16][4 * RR];        // 32 KB gate staging
  __shared__ float H2[16][RR];           //  8 KB h2 staging
  const int nBase = blockIdx.x * 16;
  const int tid   = threadIdx.x;
  const int wave  = tid >> 5;
  const int lane  = tid & 31;
  const int l15   = lane & 15;
  const bool hiL  = lane >= 16;
  const int KW    = 2 * EE + RR;         // 256

  v8f acc[8] = {};
  for (int kc = 0; kc < KW / 32; ++kc) {
    const __bf16* asrc = (kc < 4)
        ? x    + (size_t)(nBase + l15) * (2 * EE) + kc * 32
        : hrow + (size_t)(nBase + l15) * RR + (kc - 4) * 32;
    v16bf a = frag16(asrc + (hiL ? 16 : 0), asrc + (hiL ? 24 : 8));
#pragma unroll
    for (int jt = 0; jt < 8; ++jt) {
      const int j = wave * 128 + jt * 16 + l15;
      const __bf16* bp = Wcat + (size_t)j * KW + kc * 32 + (hiL ? 16 : 0);
      v16bf b = frag16(bp, bp + 8);
      acc[jt] = wmma_bf16(a, b, acc[jt]);
    }
  }
#pragma unroll
  for (int jt = 0; jt < 8; ++jt) {
    const int j = wave * 128 + jt * 16 + l15;
    const float bsum = b_ih[j] + b_hh[j];
#pragma unroll
    for (int i = 0; i < 8; ++i)
      G[i + (hiL ? 8 : 0)][j] = acc[jt][i] + bsum;
  }
  __syncthreads();

  {   // LSTM pointwise: 16 r-values per thread
    const int nl = tid >> 3;
    const int rb = (tid & 7) * 16;
    const int n  = nBase + nl;
#pragma unroll
    for (int rr = 0; rr < 16; ++rr) {
      const int r = rb + rr;
      const float ig = G[nl][r];
      const float fg = G[nl][RR + r];
      const float gg = G[nl][2 * RR + r];
      const float og = G[nl][3 * RR + r];
      const float cold = c[(size_t)n * RR + r];
      const float c2 = sigm(fg) * cold + sigm(ig) * tanhf(gg);
      const float h2 = sigm(og) * tanhf(c2);
      c[(size_t)n * RR + r] = c2;
      hrow_new[(size_t)n * RR + r] = (__bf16)h2;
      hT_new[(size_t)r * NN + n]   = (__bf16)h2;
      H2[nl][r] = h2;
      if (h_final) {
        h_final[(size_t)n * RR + r] = h2;
        c_final[(size_t)n * RR + r] = c2;
      }
    }
  }
  __syncthreads();
  if (tid < 16 * OO) {   // out = h2 @ W_out^T + b_out
    const int nl = tid / OO, o = tid % OO;
    float s = b_out[o];
#pragma unroll 4
    for (int r = 0; r < RR; ++r) s += H2[nl][r] * W_out[o * RR + r];
    out_t[(size_t)(nBase + nl) * OO + o] = s;
  }
}

// ---------------------------------------------------------------------------
extern "C" void kernel_launch(void* const* d_in, const int* in_sizes, int n_in,
                              void* d_out, int out_size, void* d_ws, size_t ws_size,
                              hipStream_t stream) {
  (void)in_sizes; (void)n_in; (void)out_size; (void)ws_size;
  const float* nodes    = (const float*)d_in[0];
  const float* grids    = (const float*)d_in[1];
  const float* hidden0  = (const float*)d_in[2];
  const float* cell0    = (const float*)d_in[3];
  const float* W_in     = (const float*)d_in[4];
  const float* b_in     = (const float*)d_in[5];
  const float* W_tensor = (const float*)d_in[6];
  const float* b_tensor = (const float*)d_in[7];
  const float* W_ih     = (const float*)d_in[8];
  const float* b_ih     = (const float*)d_in[9];
  const float* W_hh     = (const float*)d_in[10];
  const float* b_hh     = (const float*)d_in[11];
  const float* W_out    = (const float*)d_in[12];
  const float* b_out    = (const float*)d_in[13];

  char* ws = (char*)d_ws;                       // ~3.2 MB total
  float*  c_ws   = (float*) (ws + 0);           // 256 KB
  __bf16* hrow   = (__bf16*)(ws + 262144);      // 128 KB  h [N,R]
  __bf16* hT     = (__bf16*)(ws + 393216);      // 128 KB  h^T [R,N]
  __bf16* x      = (__bf16*)(ws + 524288);      // 128 KB  [ie|te]
  __bf16* social = (__bf16*)(ws + 655360);      //   2 MB  [N, G2*R]
  __bf16* Wcat   = (__bf16*)(ws + 2752512);     // 256 KB  [W_ih|W_hh]
  __bf16* Wt     = (__bf16*)(ws + 3014656);     // 256 KB  W_tensor

  float* outp    = (float*)d_out;               // outputs[T,N,O] + h_f + c_f
  float* h_final = outp + (size_t)TT * NN * OO;
  float* c_final = h_final + (size_t)NN * RR;

  init_kernel<<<512, 256, 0, stream>>>(hidden0, cell0, W_ih, W_hh, W_tensor,
                                       c_ws, hrow, hT, Wcat, Wt);
  for (int t = 0; t < TT; ++t) {
    social_wmma_kernel<<<NN / 2, 256, 0, stream>>>(
        grids + (size_t)t * NN * NN * G2C, hT, social);
    embed_kernel<<<NN / 16, 128, 0, stream>>>(
        social, Wt, b_tensor, nodes + (size_t)t * NN * II, W_in, b_in, x);
    const bool last = (t == TT - 1);
    lstm_kernel<<<NN / 16, 128, 0, stream>>>(
        x, hrow, Wcat, b_ih, b_hh, c_ws, W_out, b_out,
        outp + (size_t)t * NN * OO, hrow, hT,
        last ? h_final : nullptr, last ? c_final : nullptr);
  }
}
